// GATLayer_74302934220892
// MI455X (gfx1250) — matrix operand order repaired and checked by reading, not compiled, not run on hardware
//
#include <hip/hip_runtime.h>
#include <hip/hip_bf16.h>

// ---------------------------------------------------------------------------
// Problem constants (from the reference)
// ---------------------------------------------------------------------------
constexpr int   N_NODES = 50000;
constexpr int   N_EDGES = 500000;
constexpr int   D       = 128;            // D_IN == D_OUT == D_REL
constexpr int   DW      = 2 * D + D;      // 384 concat width
constexpr int   KCHUNKS = DW / 32;        // 12 chunks of K=32 (bf16 WMMA)
constexpr int   K_TOP   = N_EDGES / 2;    // RATIO = 0.5
constexpr float NEG_SLOPE = 0.2f;
constexpr float LN_EPS    = 1e-5f;

// ---------------------------------------------------------------------------
// WMMA fragment types (CDNA5 gfx1250, wave32)
// ---------------------------------------------------------------------------
typedef __attribute__((ext_vector_type(16))) __bf16 v16bf;
typedef __attribute__((ext_vector_type(8)))  float  v8f;

union BF16Frag {
    v16bf          v;
    unsigned short u[16];
    uint4          q[2];
};

__device__ inline unsigned short f2bf(float f) {
    unsigned int u = __float_as_uint(f);
    unsigned int r = u + 0x7FFFu + ((u >> 16) & 1u);   // round-to-nearest-even
    return (unsigned short)(r >> 16);
}

// A-fragment (16x32 bf16, MxK): lane L -> row M = L&15; K = kbase..kbase+7 in
// VGPR0..3 and kbase+16..kbase+23 in VGPR4..7, where kbase = chunk*32 + (L>=16)*8
__device__ inline v16bf loadA_bf16(const unsigned short* __restrict__ row, int kbase) {
    BF16Frag f;
    f.q[0] = *reinterpret_cast<const uint4*>(row + kbase);
    f.q[1] = *reinterpret_cast<const uint4*>(row + kbase + 16);
    return f.v;
}

__device__ inline v16bf loadA_f32(const float* __restrict__ row, int kbase) {
    BF16Frag f;
#pragma unroll
    for (int j = 0; j < 8; ++j) f.u[j]     = f2bf(row[kbase + j]);
#pragma unroll
    for (int j = 0; j < 8; ++j) f.u[8 + j] = f2bf(row[kbase + 16 + j]);
    return f.v;
}

// B-fragment (32x16 bf16, KxN) from LDS column-major W: lane L -> col N = L&15,
// K = kloc..kloc+7 and kloc+16..kloc+23, kloc = (L>=16 ? 8 : 0) (matches A)
__device__ inline v16bf loadB_lds(const unsigned short* __restrict__ bs, int kloc) {
    BF16Frag f;
    f.q[0] = *reinterpret_cast<const uint4*>(bs + kloc);
    f.q[1] = *reinterpret_cast<const uint4*>(bs + kloc + 16);
    return f.v;
}

__device__ inline void atomicMaxF(float* addr, float val) {
    int cur = __float_as_int(*addr);
    while (__int_as_float(cur) < val) {
        int old = atomicCAS(reinterpret_cast<int*>(addr), cur, __float_as_int(val));
        if (old == cur) break;
        cur = old;
    }
}

// ---------------------------------------------------------------------------
// Prep: Wt[i][n][k] = bf16(W[i][k][n])  (column-major W for contiguous-K B frags)
// ---------------------------------------------------------------------------
__global__ void prep_w_kernel(const float* __restrict__ W, unsigned short* __restrict__ Wt) {
    const int total = 4 * D * DW;
    for (int idx = blockIdx.x * blockDim.x + threadIdx.x; idx < total;
         idx += gridDim.x * blockDim.x) {
        int i   = idx / (D * DW);
        int rem = idx - i * (D * DW);
        int n   = rem / DW;
        int k   = rem - n * DW;
        Wt[idx] = f2bf(W[((size_t)i * DW + k) * D + n]);
    }
}

// ---------------------------------------------------------------------------
// xc = x @ xw + xb   (1.6 GFLOP, negligible vs edge GEMMs; plain FMA loop)
// ---------------------------------------------------------------------------
__global__ void xlinear_kernel(const float* __restrict__ x, const float* __restrict__ xw,
                               const float* __restrict__ xb, float* __restrict__ out) {
    const int total = N_NODES * D;
    for (int idx = blockIdx.x * blockDim.x + threadIdx.x; idx < total;
         idx += gridDim.x * blockDim.x) {
        const int n = idx >> 7, j = idx & (D - 1);
        float s = xb[j];
        const float* xr = x + (size_t)n * D;
#pragma unroll 8
        for (int k = 0; k < D; ++k) s = fmaf(xr[k], xw[k * D + j], s);
        out[idx] = s;
    }
}

// Per-iteration init: bf16 copy of current features + zero the accumulation buffer
__global__ void pack_zero_kernel(const float* __restrict__ cur,
                                 unsigned short* __restrict__ xcb,
                                 float* __restrict__ xcNext) {
    const int total = N_NODES * D;
    for (int i = blockIdx.x * blockDim.x + threadIdx.x; i < total;
         i += gridDim.x * blockDim.x) {
        xcb[i]    = f2bf(cur[i]);
        xcNext[i] = 0.f;
    }
}

__global__ void reset_seg_kernel(float* __restrict__ segmax, float* __restrict__ segden,
                                 unsigned int* __restrict__ hist) {
    for (int i = blockIdx.x * blockDim.x + threadIdx.x; i < N_NODES;
         i += gridDim.x * blockDim.x) {
        segmax[i] = -1e30f;
        segden[i] = 0.f;
        if (i < 1024) hist[i] = 0u;
    }
}

// ---------------------------------------------------------------------------
// Core: one wave32 computes edge_h[16,128] = [16,384]bf16 @ Wt via 96 WMMAs.
// PHASE 0: score = leaky_relu(edge_h @ a), + segment max.
// PHASE 1: recompute edge_h, scatter-add masked alpha * edge_h to xcNext.
// ---------------------------------------------------------------------------
template <int PHASE>
__global__ __launch_bounds__(256)
void edge_gemm_kernel(const int* __restrict__ src, const int* __restrict__ dst,
                      const unsigned short* __restrict__ xcb,
                      const float* __restrict__ ete,
                      const unsigned short* __restrict__ Wt,
                      const float* __restrict__ avec,      // PHASE 0
                      float* __restrict__ score,           // PHASE 0
                      float* __restrict__ segmax,          // PHASE 0
                      const float* __restrict__ alpha,     // PHASE 1
                      const float* __restrict__ threshp,   // PHASE 1
                      float* __restrict__ xcNext)          // PHASE 1
{
    // B staged in LDS, pitch 40 ushorts (80 B) -> 16 B aligned, conflict-free b128
    __shared__ unsigned short Bs[D * 40];

    const int  lane = threadIdx.x & 31;
    const int  wave = threadIdx.x >> 5;
    const int  half = lane >> 4;            // 0: K 0-7/16-23 rows 0-7 ; 1: K 8-15/24-31 rows 8-15
    const int  l16  = lane & 15;
    const long long base = ((long long)blockIdx.x * 8 + wave) * 16;

    long long eRow = base + l16;
    if (eRow >= N_EDGES) eRow = N_EDGES - 1;
    const int sN = src[eRow];
    const int dN = dst[eRow];
    const unsigned short* srow = xcb + (size_t)sN * D;
    const unsigned short* drow = xcb + (size_t)dN * D;
    const float*          erow = ete + (size_t)eRow * D;
    __builtin_prefetch(erow, 0, 1);   // global_prefetch_b8

    v8f acc[8] = {};
    const int kb   = half * 8;
    const int kloc = half * 8;

    for (int c = 0; c < KCHUNKS; ++c) {
        __syncthreads();
        {   // cooperative stage of Wt chunk c: 128 cols x 32 K (8 KB)
            const int col = threadIdx.x >> 1;
            const int h16 = (threadIdx.x & 1) * 16;
            const uint4* g = reinterpret_cast<const uint4*>(Wt + (size_t)col * DW + c * 32 + h16);
            uint4*       s = reinterpret_cast<uint4*>(Bs + col * 40 + h16);
            s[0] = g[0];
            s[1] = g[1];
        }
        __syncthreads();

        v16bf afrag;
        if (c < 4)      afrag = loadA_bf16(srow, c * 32 + kb);
        else if (c < 8) afrag = loadA_bf16(drow, (c - 4) * 32 + kb);
        else            afrag = loadA_f32 (erow, (c - 8) * 32 + kb);

#pragma unroll
        for (int t = 0; t < 8; ++t) {
            v16bf bfrag = loadB_lds(Bs + (t * 16 + l16) * 40, kloc);
            acc[t] = __builtin_amdgcn_wmma_f32_16x16x32_bf16(
                false, afrag, false, bfrag, (short)0, acc[t], false, false);
        }
    }

    // D layout: lane holds col t*16 + l16, rows half*8 + r (r=0..7) in acc[t][r]
    if (PHASE == 0) {
        float part[8];
#pragma unroll
        for (int r = 0; r < 8; ++r) part[r] = 0.f;
#pragma unroll
        for (int t = 0; t < 8; ++t) {
            const float aw = avec[t * 16 + l16];
#pragma unroll
            for (int r = 0; r < 8; ++r) part[r] = fmaf(acc[t][r], aw, part[r]);
        }
#pragma unroll
        for (int m = 1; m <= 8; m <<= 1) {
#pragma unroll
            for (int r = 0; r < 8; ++r) part[r] += __shfl_xor(part[r], m, 32);
        }
        if (l16 == 0) {
#pragma unroll
            for (int r = 0; r < 8; ++r) {
                const long long e = base + half * 8 + r;
                if (e < N_EDGES) {
                    float s = part[r];
                    s = s > 0.f ? s : NEG_SLOPE * s;     // LeakyReLU
                    score[e] = s;
                    atomicMaxF(&segmax[src[e]], s);
                }
            }
        }
    } else {
        const float th = *threshp;
        float al[8];
        int   sr[8];
#pragma unroll
        for (int r = 0; r < 8; ++r) {
            const long long e = base + half * 8 + r;
            if (e < N_EDGES) {
                const float a_ = alpha[e];
                al[r] = (a_ >= th) ? a_ : 0.f;           // top-k mask
                sr[r] = src[e];
            } else { al[r] = 0.f; sr[r] = 0; }
        }
#pragma unroll
        for (int t = 0; t < 8; ++t) {
            const int cg = t * 16 + l16;
#pragma unroll
            for (int r = 0; r < 8; ++r) {
                if (al[r] != 0.f)
                    atomicAdd(&xcNext[(size_t)sr[r] * D + cg], al[r] * acc[t][r]);
            }
        }
    }
}

// ---------------------------------------------------------------------------
// Segment softmax pieces
// ---------------------------------------------------------------------------
__global__ void exp_kernel(const int* __restrict__ src, const float* __restrict__ score,
                           const float* __restrict__ segmax, float* __restrict__ eexp,
                           float* __restrict__ segden) {
    for (int e = blockIdx.x * blockDim.x + threadIdx.x; e < N_EDGES;
         e += gridDim.x * blockDim.x) {
        const int s = src[e];
        const float v = __expf(score[e] - segmax[s]);
        eexp[e] = v;
        atomicAdd(&segden[s], v);
    }
}

__global__ void alpha_kernel(const int* __restrict__ src, const float* __restrict__ eexp,
                             const float* __restrict__ segden, float* __restrict__ alpha,
                             unsigned int* __restrict__ hist) {
    for (int e = blockIdx.x * blockDim.x + threadIdx.x; e < N_EDGES;
         e += gridDim.x * blockDim.x) {
        const float a = eexp[e] / (segden[src[e]] + 1e-16f);
        alpha[e] = a;
        int b = (int)(a * 1024.f);
        b = b < 0 ? 0 : (b > 1023 ? 1023 : b);
        atomicAdd(&hist[b], 1u);
    }
}

// Radix-select style threshold: walk histogram from the top until k edges kept
__global__ void thresh_kernel(const unsigned int* __restrict__ hist,
                              float* __restrict__ thresh) {
    if (threadIdx.x == 0 && blockIdx.x == 0) {
        unsigned int acc = 0;
        float th = 0.f;
        for (int b = 1023; b >= 0; --b) {
            acc += hist[b];
            if (acc >= (unsigned int)K_TOP) { th = (float)b / 1024.f; break; }
        }
        thresh[0] = th;
    }
}

// ---------------------------------------------------------------------------
// ELU + LayerNorm, one row per 128-thread block (in-place on d_out)
// ---------------------------------------------------------------------------
__global__ void elu_ln_kernel(float* __restrict__ cur, const float* __restrict__ gamma,
                              const float* __restrict__ beta) {
    __shared__ float redS[4], redQ[4];
    const int row = blockIdx.x, j = threadIdx.x, wv = threadIdx.x >> 5;
    float v = cur[(size_t)row * D + j];
    v = v > 0.f ? v : expm1f(v);                       // ELU
    float s = v, q = v * v;
#pragma unroll
    for (int m = 1; m <= 16; m <<= 1) {
        s += __shfl_xor(s, m, 32);
        q += __shfl_xor(q, m, 32);
    }
    if ((threadIdx.x & 31) == 0) { redS[wv] = s; redQ[wv] = q; }
    __syncthreads();
    float S = redS[0] + redS[1] + redS[2] + redS[3];
    float Q = redQ[0] + redQ[1] + redQ[2] + redQ[3];
    const float mu  = S * (1.f / D);
    const float var = Q * (1.f / D) - mu * mu;
    cur[(size_t)row * D + j] = (v - mu) * rsqrtf(var + LN_EPS) * gamma[j] + beta[j];
}

// ---------------------------------------------------------------------------
// Host launcher
// ---------------------------------------------------------------------------
extern "C" void kernel_launch(void* const* d_in, const int* in_sizes, int n_in,
                              void* d_out, int out_size, void* d_ws, size_t ws_size,
                              hipStream_t stream) {
    (void)in_sizes; (void)n_in; (void)out_size; (void)ws_size;

    const int*   eidx  = (const int*)d_in[0];
    const int*   src   = eidx;
    const int*   dst   = eidx + N_EDGES;
    const float* x     = (const float*)d_in[1];
    const float* ete   = (const float*)d_in[2];
    const float* W     = (const float*)d_in[3];
    const float* a     = (const float*)d_in[4];
    const float* xw    = (const float*)d_in[5];
    const float* xb    = (const float*)d_in[6];
    const float* gamma = (const float*)d_in[7];
    const float* beta  = (const float*)d_in[8];
    float*       out   = (float*)d_out;

    // Workspace carve-up (~45 MB)
    char* ws = (char*)d_ws;
    auto up = [](size_t v) { return (v + 255) & ~size_t(255); };
    size_t off = 0;
    unsigned short* Wt  = (unsigned short*)(ws + off); off += up((size_t)4 * D * DW * 2);
    unsigned short* xcb = (unsigned short*)(ws + off); off += up((size_t)N_NODES * D * 2);
    float* xcB    = (float*)(ws + off); off += up((size_t)N_NODES * D * 4);
    float* score  = (float*)(ws + off); off += up((size_t)N_EDGES * 4);
    float* eexp   = (float*)(ws + off); off += up((size_t)N_EDGES * 4);
    float* alpha  = (float*)(ws + off); off += up((size_t)N_EDGES * 4);
    float* segmax = (float*)(ws + off); off += up((size_t)N_NODES * 4);
    float* segden = (float*)(ws + off); off += up((size_t)N_NODES * 4);
    unsigned int* hist = (unsigned int*)(ws + off); off += up(1024 * 4);
    float* thresh = (float*)(ws + off); off += up(256);

    const int edgeBlocks = (N_EDGES / 16 + 7) / 8;   // 8 wave-tiles per 256-thread block

    prep_w_kernel<<<192, 256, 0, stream>>>(W, Wt);
    xlinear_kernel<<<(N_NODES * D + 255) / 256, 256, 0, stream>>>(x, xw, xb, out);

    float* cur = out;
    float* nxt = xcB;
    for (int i = 0; i < 4; ++i) {
        const unsigned short* Wti = Wt + (size_t)i * D * DW;
        const float*          ai  = a + (size_t)i * D;

        pack_zero_kernel<<<(N_NODES * D + 255) / 256, 256, 0, stream>>>(cur, xcb, nxt);
        reset_seg_kernel<<<(N_NODES + 255) / 256, 256, 0, stream>>>(segmax, segden, hist);

        edge_gemm_kernel<0><<<edgeBlocks, 256, 0, stream>>>(
            src, dst, xcb, ete, Wti, ai, score, segmax, nullptr, nullptr, nullptr);

        exp_kernel<<<1954, 256, 0, stream>>>(src, score, segmax, eexp, segden);
        alpha_kernel<<<1954, 256, 0, stream>>>(src, eexp, segden, alpha, hist);
        thresh_kernel<<<1, 32, 0, stream>>>(hist, thresh);

        edge_gemm_kernel<1><<<edgeBlocks, 256, 0, stream>>>(
            src, dst, xcb, ete, Wti, nullptr, nullptr, nullptr, alpha, thresh, nxt);

        float* t = cur; cur = nxt; nxt = t;   // ends with cur == out after 4 swaps
    }

    elu_ln_kernel<<<N_NODES, 128, 0, stream>>>(out, gamma, beta);
}
